// GraphAttention_12214886990524
// MI455X (gfx1250) — matrix operand order repaired and checked by reading, not compiled, not run on hardware
//
#include <hip/hip_runtime.h>

// ---------------- types ----------------
typedef __attribute__((ext_vector_type(16))) __bf16 v16bf;
typedef __attribute__((ext_vector_type(8)))  float  v8f;
typedef int vi4 __attribute__((vector_size(16)));   // matches async builtin param type

#define BB   16
#define NN   2048
#define FIN  256
#define FOUT 128
#define LRELU 0.2f
#define NEG_BIG -9.0e15f

#if __has_builtin(__builtin_amdgcn_global_load_async_to_lds_b128)
#define HAVE_ASYNC_LDS 1
#endif

// ---------------- helpers ----------------
// round f32 for bf16 truncation: +0x8000 = round-to-nearest (ties away)
__device__ __forceinline__ unsigned bfround(float x) {
  return __builtin_bit_cast(unsigned, x) + 0x8000u;
}
// two f32 -> packed bf16 pair (x in low half) : 2x v_add + 1x v_perm_b32
__device__ __forceinline__ unsigned pack2bf(float x, float y) {
  return __builtin_amdgcn_perm(bfround(y), bfround(x), 0x07060302u);
}

// 16-byte global -> LDS copy; async on CDNA5 (tracked by ASYNCcnt)
__device__ __forceinline__ void g2l_b128(const void* g, void* l) {
#if defined(HAVE_ASYNC_LDS)
  typedef __attribute__((address_space(1))) vi4 gvi4;
  typedef __attribute__((address_space(3))) vi4 lvi4;
  void* gnc = const_cast<void*>(g);
  __builtin_amdgcn_global_load_async_to_lds_b128(
      (gvi4*)gnc, (lvi4*)l, /*offset=*/0, /*cpol=*/0);
#else
  *(uint4*)l = *(const uint4*)g;
#endif
}

template <int N>
__device__ __forceinline__ void wait_async() {
#if defined(HAVE_ASYNC_LDS)
#if __has_builtin(__builtin_amdgcn_s_wait_asynccnt)
  __builtin_amdgcn_s_wait_asynccnt(N);
#else
  asm volatile("s_wait_asynccnt %0" ::"i"(N) : "memory");
#endif
#endif
}

// =======================================================================
// Kernel 0: W (f32) -> W (bf16), done once.
// =======================================================================
__global__ __launch_bounds__(256) void w_cvt_kernel(
    const float* __restrict__ W, unsigned* __restrict__ wbf) {
  const int i = blockIdx.x * 256 + threadIdx.x;  // pair index
  const float2 t = *(const float2*)(W + 2 * i);
  wbf[i] = pack2bf(t.x, t.y);
}

// =======================================================================
// Kernel 1: h[b,n,:] = x[b,n,:] @ W   (bf16 WMMA, f32 accum, bf16 store)
// One wave handles 16 nodes x 128 features. 8 waves/block -> 128 nodes.
// =======================================================================
__global__ __launch_bounds__(256) void h_gemm_kernel(
    const float* __restrict__ inp, const unsigned short* __restrict__ wbf,
    unsigned short* __restrict__ h) {
  const int lane = threadIdx.x & 31;
  const int wave = threadIdx.x >> 5;
  const int b    = blockIdx.y;
  const int n0   = blockIdx.x * 128 + wave * 16;
  const int m    = lane & 15;
  const int hi   = lane >> 4;

  v8f zero = {0.f, 0.f, 0.f, 0.f, 0.f, 0.f, 0.f, 0.f};
  v8f acc[8];
#pragma unroll
  for (int j = 0; j < 8; ++j) acc[j] = zero;

  const float* arow = inp + ((size_t)b * NN + n0 + m) * FIN;

  for (int kk = 0; kk < FIN; kk += 32) {
    // A tile (16x32 bf16): per-lane K pattern = {0..7, 16..23} (+8 for hi half)
    const float4 q0 = *(const float4*)(arow + kk + hi * 8);
    const float4 q1 = *(const float4*)(arow + kk + hi * 8 + 4);
    const float4 q2 = *(const float4*)(arow + kk + 16 + hi * 8);
    const float4 q3 = *(const float4*)(arow + kk + 16 + hi * 8 + 4);
    union { unsigned p[8]; v16bf v; } ua;
    ua.p[0] = pack2bf(q0.x, q0.y);
    ua.p[1] = pack2bf(q0.z, q0.w);
    ua.p[2] = pack2bf(q1.x, q1.y);
    ua.p[3] = pack2bf(q1.z, q1.w);
    ua.p[4] = pack2bf(q2.x, q2.y);
    ua.p[5] = pack2bf(q2.z, q2.w);
    ua.p[6] = pack2bf(q3.x, q3.y);
    ua.p[7] = pack2bf(q3.z, q3.w);

    // B tiles: lane = feature column N, elements = K in same order as A
    const unsigned short* wrow = wbf + (size_t)kk * FOUT + m;
#pragma unroll
    for (int j = 0; j < 8; ++j) {
      union { unsigned short e[16]; v16bf v; } ub;
#pragma unroll
      for (int s = 0; s < 16; ++s)
        ub.e[s] = wrow[(size_t)((s & 7) + hi * 8 + (s >> 3) * 16) * FOUT + j * 16];
      acc[j] = __builtin_amdgcn_wmma_f32_16x16x32_bf16(
          false, ua.v, false, ub.v, (short)0, acc[j], false, false);
    }
  }

  // C layout: VGPR r, lanes0-15 -> (M=r, N=lane); lanes16-31 -> (M=r+8, N=lane-16)
  // store high-16 of rounded f32 -> global_store_d16_hi_b16 pattern
#pragma unroll
  for (int j = 0; j < 8; ++j) {
#pragma unroll
    for (int r = 0; r < 8; ++r) {
      const int row = r + 8 * hi;
      const unsigned u = bfround(acc[j][r]);
      h[((size_t)b * NN + n0 + row) * FOUT + j * 16 + m] =
          (unsigned short)(u >> 16);
    }
  }
}

// =======================================================================
// Kernel 2: f1 = h . a1, f2 = h . a2  (per node, packed bf16 pair loads)
// =======================================================================
__global__ __launch_bounds__(256) void fvec_kernel(
    const unsigned* __restrict__ h, const float* __restrict__ a1,
    const float* __restrict__ a2, float* __restrict__ f1,
    float* __restrict__ f2) {
  const int idx = blockIdx.x * 256 + threadIdx.x;  // 0 .. B*N-1
  const unsigned* hp = h + (size_t)idx * (FOUT / 2);
  float s1 = 0.f, s2 = 0.f;
#pragma unroll 8
  for (int i = 0; i < FOUT / 2; ++i) {
    const unsigned u = hp[i];
    const float lo = __builtin_bit_cast(float, u << 16);
    const float hf = __builtin_bit_cast(float, u & 0xFFFF0000u);
    s1 = fmaf(lo, a1[2 * i], s1);
    s2 = fmaf(lo, a2[2 * i], s2);
    s1 = fmaf(hf, a1[2 * i + 1], s1);
    s2 = fmaf(hf, a2[2 * i + 1], s2);
  }
  f1[idx] = s1;
  f2[idx] = s2;
}

// =======================================================================
// Kernel 3: fused masked-softmax attention + P@h + ELU (flash style).
// Block = 256 thr (8 waves), 128 rows/block; chunk = 32 columns.
// adj + h chunk double-buffered in LDS via async-to-LDS; the 256MB adj
// stream (the roofline limiter) is overlapped with WMMA/softmax compute.
// =======================================================================
__global__ __launch_bounds__(256) void gat_attn_kernel(
    const int* __restrict__ adj, const unsigned short* __restrict__ h,
    const float* __restrict__ f1g, const float* __restrict__ f2g,
    float* __restrict__ out) {
  __shared__ int            adj_lds[2][128 * 36];  // 144B rows (bank-friendly)
  __shared__ unsigned short h_lds[2][32 * 136];    // 272B rows (bank-friendly)
  __shared__ float          f2_lds[NN];

  const int tid  = threadIdx.x;
  const int lane = tid & 31;
  const int wave = tid >> 5;
  const int b    = blockIdx.y;
  const int row0 = blockIdx.x * 128;
  const int m    = lane & 15;
  const int hi   = lane >> 4;
  const int grow = row0 + wave * 16 + m;

  // stage f2 row for this batch once
  for (int i = tid; i < NN; i += 256) f2_lds[i] = f2g[(size_t)b * NN + i];
  const float f1v = f1g[(size_t)b * NN + grow];

  float m_i = -__builtin_inff();
  float l_part = 0.f;
  v8f zero = {0.f, 0.f, 0.f, 0.f, 0.f, 0.f, 0.f, 0.f};
  v8f acc[8];
#pragma unroll
  for (int j = 0; j < 8; ++j) acc[j] = zero;

  // staging assignments: adj 128x32 ints (2 thr/row), h 32x128 bf16 (8 thr/node)
  const int ra = tid >> 1;
  const int rp = (tid & 1) * 16;   // dwords
  const int hn = tid >> 3;
  const int hs = (tid & 7) * 16;   // bf16 elements (32B)

  // issue the 6 async b128 loads for one 32-column chunk into buffer bi
  auto issue_chunk = [&](int c0, int bi) {
    const int* ag = adj + ((size_t)b * NN + row0 + ra) * NN + c0 + rp;
    int* al = &adj_lds[bi][ra * 36 + rp];
    g2l_b128(ag + 0, al + 0);
    g2l_b128(ag + 4, al + 4);
    g2l_b128(ag + 8, al + 8);
    g2l_b128(ag + 12, al + 12);
    const unsigned short* hg = h + ((size_t)b * NN + c0 + hn) * FOUT + hs;
    unsigned short* hl = &h_lds[bi][hn * 136 + hs];
    g2l_b128(hg + 0, hl + 0);
    g2l_b128(hg + 8, hl + 8);
  };

  issue_chunk(0, 0);  // prologue: chunk 0 -> buffer 0

  for (int ci = 0; ci < NN / 32; ++ci) {
    const int cur = ci & 1;
    const int c0  = ci * 32;
    __syncthreads();  // everyone done computing previous chunk (frees cur^1)
    issue_chunk(((ci + 1) & 63) * 32, cur ^ 1);  // wraps to dummy on last iter
    wait_async<6>();  // only the 6 just-issued may remain -> chunk ci landed
    __syncthreads();  // chunk ci visible to all waves

    // ---- scores for 16x32 tile, computed directly in WMMA-A layout ----
    float ev[16];
    float cmax = -3.0e38f;
#pragma unroll
    for (int v = 0; v < 8; ++v) {
      const int base = (v & 3) * 2 + (v >> 2) * 16 + hi * 8;
      const float2 f2p = *(const float2*)&f2_lds[c0 + base];
      const int2   ap  = *(const int2*)&adj_lds[cur][(wave * 16 + m) * 36 + base];
      float e0 = f1v + f2p.x;
      e0 = e0 > 0.f ? e0 : LRELU * e0;
      e0 = ap.x > 0 ? e0 : NEG_BIG;
      float e1 = f1v + f2p.y;
      e1 = e1 > 0.f ? e1 : LRELU * e1;
      e1 = ap.y > 0 ? e1 : NEG_BIG;
      ev[2 * v] = e0;
      ev[2 * v + 1] = e1;
      cmax = fmaxf(cmax, fmaxf(e0, e1));
    }
    // row max lives split across lane and lane^16 (K halves)
    cmax = fmaxf(cmax, __shfl_xor(cmax, 16, 32));
    const float m_new = fmaxf(m_i, cmax);

    // rescale only when some row's max actually grew (wave-uniform skip;
    // steady-state chunks pay one compare+branch instead of 64 muls + 8 perms)
    if (__any(m_new > m_i)) {
      const float alpha = __expf(m_i - m_new);
      l_part *= alpha;
#pragma unroll
      for (int r = 0; r < 8; ++r) {
        const float ar = __shfl(alpha, r + hi * 8, 32);
#pragma unroll
        for (int j = 0; j < 8; ++j) acc[j][r] *= ar;
      }
    }
    m_i = m_new;

    float lsum = 0.f;
    union { unsigned p[8]; v16bf v; } ua;
#pragma unroll
    for (int v = 0; v < 8; ++v) {
      const float p0 = __expf(ev[2 * v] - m_new);
      const float p1 = __expf(ev[2 * v + 1] - m_new);
      lsum += p0 + p1;
      ua.p[v] = pack2bf(p0, p1);
    }
    l_part += lsum;

    // P (16x32) @ h-chunk (32x16 per feature tile): lane = feature N, elements = K (node in chunk)
#pragma unroll
    for (int j = 0; j < 8; ++j) {
      union { unsigned short e[16]; v16bf v; } ub;
#pragma unroll
      for (int s = 0; s < 16; ++s)
        ub.e[s] = h_lds[cur][((s & 7) + hi * 8 + (s >> 3) * 16) * 136 + j * 16 + m];
      acc[j] = __builtin_amdgcn_wmma_f32_16x16x32_bf16(
          false, ua.v, false, ub.v, (short)0, acc[j], false, false);
    }
  }

  // normalize + ELU + store
  const float ltot = l_part + __shfl_xor(l_part, 16, 32);
#pragma unroll
  for (int r = 0; r < 8; ++r) {
    const float linv = 1.f / __shfl(ltot, r + hi * 8, 32);
    const int row = row0 + wave * 16 + r + 8 * hi;
    float* orow = out + ((size_t)b * NN + row) * FOUT;
#pragma unroll
    for (int j = 0; j < 8; ++j) {
      float x = acc[j][r] * linv;
      x = x > 0.f ? x : (__expf(x) - 1.f);
      orow[j * 16 + m] = x;
    }
  }
}

// =======================================================================
extern "C" void kernel_launch(void* const* d_in, const int* in_sizes, int n_in,
                              void* d_out, int out_size, void* d_ws,
                              size_t ws_size, hipStream_t stream) {
  (void)in_sizes; (void)n_in; (void)out_size; (void)ws_size;
  const float* inp = (const float*)d_in[0];   // [B,N,FIN]
  const int*   adj = (const int*)d_in[1];     // [B,N,N]
  const float* W   = (const float*)d_in[2];   // [FIN,FOUT]
  const float* a1  = (const float*)d_in[3];   // [FOUT,1]
  const float* a2  = (const float*)d_in[4];   // [FOUT,1]
  float* out = (float*)d_out;                 // [B,N,FOUT]

  // workspace: h (bf16, 8MB) | f1 (128KB) | f2 (128KB) | Wbf16 (64KB)
  unsigned short* hws = (unsigned short*)d_ws;
  const size_t hbytes = (size_t)BB * NN * FOUT * sizeof(unsigned short);
  float* f1 = (float*)((char*)d_ws + hbytes);
  float* f2 = f1 + (size_t)BB * NN;
  unsigned* wbf = (unsigned*)(f2 + (size_t)BB * NN);

  w_cvt_kernel<<<dim3((FIN * FOUT) / 512), 256, 0, stream>>>(W, wbf);
  h_gemm_kernel<<<dim3(NN / 128, BB), 256, 0, stream>>>(
      inp, (const unsigned short*)wbf, hws);
  fvec_kernel<<<dim3((BB * NN) / 256), 256, 0, stream>>>(
      (const unsigned*)hws, a1, a2, f1, f2);
  gat_attn_kernel<<<dim3(NN / 128, BB), 256, 0, stream>>>(adj, hws, f1, f2, out);
}
